// MultiheadSelfAttention_54460185313804
// MI455X (gfx1250) — matrix operand order repaired and checked by reading, not compiled, not run on hardware
//
#include <hip/hip_runtime.h>

#define D_MODEL 1024
#define SEQ     2048
#define BATCH   2
#define NHEAD   16
#define HDIM    64
#define MTOT    (BATCH*SEQ)   // 4096 rows

typedef __attribute__((ext_vector_type(16))) __bf16 v16bf;
typedef __attribute__((ext_vector_type(8)))  float  v8f;

union Frag { uint4 q[2]; v16bf v; };

// Load a 16-element bf16 fragment as two 16B chunks (p0[0..7], p1[0..7]).
__device__ __forceinline__ v16bf ld_frag(const unsigned short* p0, const unsigned short* p1) {
  Frag f;
  f.q[0] = *(const uint4*)p0;
  f.q[1] = *(const uint4*)p1;
  return f.v;
}

__device__ __forceinline__ v8f wmma_bf16(v16bf a, v16bf b, v8f c) {
  // D = A(16x32 bf16) * B(32x16 bf16) + C(16x16 f32)
  return __builtin_amdgcn_wmma_f32_16x16x32_bf16(false, a, false, b, (short)0, c, false, false);
}

__device__ __forceinline__ unsigned short bf16_rn(float f) {
  unsigned u = __float_as_uint(f);
  u += 0x7FFFu + ((u >> 16) & 1u);      // round-to-nearest-even
  return (unsigned short)(u >> 16);
}
__device__ __forceinline__ float bf16_f32(unsigned short h) {
  return __uint_as_float(((unsigned)h) << 16);
}

// ---------------------------------------------------------------------------
// Split fp32 -> (hi, lo) bf16 planes:  x ~= hi + lo  (bf16x3 GEMM scheme)
// ---------------------------------------------------------------------------
__global__ void split_kernel(const float* __restrict__ src,
                             unsigned short* __restrict__ hi,
                             unsigned short* __restrict__ lo, int n) {
  int i = blockIdx.x * blockDim.x + threadIdx.x;
  if (i < n) {
    float x = src[i];
    unsigned short h = bf16_rn(x);
    hi[i] = h;
    lo[i] = bf16_rn(x - bf16_f32(h));
  }
}

// ---------------------------------------------------------------------------
// C[M,N] = (Ahi+Alo)[M,K] * ((Bhi+Blo)[N,K])^T + bias[N]   (3-product bf16x3)
// Block: 256 threads = 8 waves (2M x 4N); block tile 64x128, wave tile 32x32.
// Each wave: 2x2 16x16 C tiles -> 12 WMMAs per 16 b128 loads per k-step.
// MODE 0: write split bf16 planes OutHi/OutLo [M,N]              (Q, K)
// MODE 1: write split planes transposed per head Vt[b][h][d][s]  (V)
// MODE 2: write fp32 OutF [M,N]                                  (final out)
// ---------------------------------------------------------------------------
template<int MODE>
__global__ __launch_bounds__(256) void gemm_bf16x3(
    const unsigned short* __restrict__ Ahi, const unsigned short* __restrict__ Alo,
    const unsigned short* __restrict__ Bhi, const unsigned short* __restrict__ Blo,
    const float* __restrict__ bias,
    unsigned short* __restrict__ OutHi, unsigned short* __restrict__ OutLo,
    float* __restrict__ OutF, int N, int K)
{
  const int lane = threadIdx.x & 31;
  const int wave = threadIdx.x >> 5;
  const int wm = wave & 1;
  const int wn = wave >> 1;
  const int m0 = blockIdx.x * 64  + wm * 32;
  const int n0 = blockIdx.y * 128 + wn * 32;
  const int col  = lane & 15;
  const int half = lane >> 4;

  const size_t aRow0 = (size_t)(m0 + col) * K;        // A: row = M = lane&15
  const size_t aRow1 = aRow0 + (size_t)16 * K;
  const size_t bRow0 = (size_t)(n0 + col) * K;        // B: col = N = lane&15
  const size_t bRow1 = bRow0 + (size_t)16 * K;

  v8f c00 = {0.f,0.f,0.f,0.f,0.f,0.f,0.f,0.f};
  v8f c01 = c00, c10 = c00, c11 = c00;

  for (int k0 = 0; k0 < K; k0 += 32) {
    const size_t a0 = aRow0 + k0 + 8 * half;          // A chunks: k+8h, k+16+8h
    const size_t a1 = aRow1 + k0 + 8 * half;
    v16bf aH0 = ld_frag(Ahi + a0, Ahi + a0 + 16);
    v16bf aL0 = ld_frag(Alo + a0, Alo + a0 + 16);
    v16bf aH1 = ld_frag(Ahi + a1, Ahi + a1 + 16);
    v16bf aL1 = ld_frag(Alo + a1, Alo + a1 + 16);
    const size_t b0 = bRow0 + k0 + 8 * half;          // B chunks: k+8h, k+16+8h
    const size_t b1 = bRow1 + k0 + 8 * half;
    v16bf b0H = ld_frag(Bhi + b0, Bhi + b0 + 16);
    v16bf b0L = ld_frag(Blo + b0, Blo + b0 + 16);
    v16bf b1H = ld_frag(Bhi + b1, Bhi + b1 + 16);
    v16bf b1L = ld_frag(Blo + b1, Blo + b1 + 16);

    c00 = wmma_bf16(aH0, b0H, c00);
    c00 = wmma_bf16(aH0, b0L, c00);
    c00 = wmma_bf16(aL0, b0H, c00);
    c01 = wmma_bf16(aH0, b1H, c01);
    c01 = wmma_bf16(aH0, b1L, c01);
    c01 = wmma_bf16(aL0, b1H, c01);
    c10 = wmma_bf16(aH1, b0H, c10);
    c10 = wmma_bf16(aH1, b0L, c10);
    c10 = wmma_bf16(aL1, b0H, c10);
    c11 = wmma_bf16(aH1, b1H, c11);
    c11 = wmma_bf16(aH1, b1L, c11);
    c11 = wmma_bf16(aL1, b1H, c11);
  }

  const float bn0 = bias[n0 + col];
  const float bn1 = bias[n0 + 16 + col];

  v8f cm[2][2] = {{c00, c01}, {c10, c11}};
  #pragma unroll
  for (int mt = 0; mt < 2; ++mt) {
    #pragma unroll
    for (int r = 0; r < 8; ++r) {
      const int mrow = m0 + mt * 16 + r + 8 * half;   // C layout: M = r + 8*half
      float v0 = cm[mt][0][r] + bn0;
      float v1 = cm[mt][1][r] + bn1;
      if (MODE == 0) {
        size_t i0 = (size_t)mrow * N + n0 + col;
        size_t i1 = i0 + 16;
        unsigned short h0 = bf16_rn(v0);
        OutHi[i0] = h0; OutLo[i0] = bf16_rn(v0 - bf16_f32(h0));
        unsigned short h1 = bf16_rn(v1);
        OutHi[i1] = h1; OutLo[i1] = bf16_rn(v1 - bf16_f32(h1));
      } else if (MODE == 1) {
        const int b = mrow >> 11;                     // row = b*SEQ + s
        const int s = mrow & (SEQ - 1);
        const int na = n0 + col, nb = n0 + 16 + col;  // n = h*64 + d
        size_t i0 = ((size_t)((b * NHEAD + (na >> 6)) * HDIM + (na & 63))) * SEQ + s;
        size_t i1 = ((size_t)((b * NHEAD + (nb >> 6)) * HDIM + (nb & 63))) * SEQ + s;
        unsigned short h0 = bf16_rn(v0);
        OutHi[i0] = h0; OutLo[i0] = bf16_rn(v0 - bf16_f32(h0));
        unsigned short h1 = bf16_rn(v1);
        OutHi[i1] = h1; OutLo[i1] = bf16_rn(v1 - bf16_f32(h1));
      } else {
        OutF[(size_t)mrow * N + n0 + col]      = v0;
        OutF[(size_t)mrow * N + n0 + 16 + col] = v1;
      }
    }
  }
}

// ---------------------------------------------------------------------------
// Flash attention: one wave = one 16-row Q tile for one (b,h); key chunks of
// 64 with online softmax (amortizes the shuffle/exp overhead over 48 WMMAs
// per chunk).  Scores use bf16x3 WMMA; P is staged through LDS to re-layout
// the 16x16 C tiles into 16x32 A operands for the PV WMMAs.
// Writes split bf16 O planes [MTOT, D_MODEL] for the final projection.
// ---------------------------------------------------------------------------
__global__ __launch_bounds__(256) void flash_attn(
    const unsigned short* __restrict__ Qhi, const unsigned short* __restrict__ Qlo,
    const unsigned short* __restrict__ Khi, const unsigned short* __restrict__ Klo,
    const unsigned short* __restrict__ Vthi, const unsigned short* __restrict__ Vtlo,
    const int* __restrict__ amask,
    unsigned short* __restrict__ Ohi, unsigned short* __restrict__ Olo)
{
  __shared__ unsigned short ldsP[8][2][16 * 64];     // per-wave P tile (hi,lo)
  const int lane = threadIdx.x & 31;
  const int wave = threadIdx.x >> 5;
  const int gw = blockIdx.x * 8 + wave;
  const int qt = gw & 127;                           // 128 q-tiles per (b,h)
  const int bh = gw >> 7;
  const int b  = bh >> 4;
  const int h  = bh & 15;
  const int col = lane & 15, half = lane >> 4;

  // Q A-fragments: 16 rows x 64 d  -> two 32-deep k-steps, hi+lo planes
  v16bf qH[2], qL[2];
  {
    const size_t qr = (size_t)(b * SEQ + qt * 16 + col) * D_MODEL + h * HDIM;
    #pragma unroll
    for (int t = 0; t < 2; ++t) {
      size_t o = qr + t * 32 + 8 * half;
      qH[t] = ld_frag(Qhi + o, Qhi + o + 16);
      qL[t] = ld_frag(Qlo + o, Qlo + o + 16);
    }
  }

  float mrun[8], lrun[8];
  #pragma unroll
  for (int r = 0; r < 8; ++r) { mrun[r] = -1e30f; lrun[r] = 0.f; }
  v8f oacc[4];
  #pragma unroll
  for (int dt = 0; dt < 4; ++dt)
    oacc[dt] = (v8f){0.f,0.f,0.f,0.f,0.f,0.f,0.f,0.f};

  const float scale = 0.125f;                        // 1/sqrt(64)

  for (int n0 = 0; n0 < SEQ; n0 += 64) {
    // ---- scores: four 16-col tiles over this 64-key chunk --------------
    v8f s[4];
    #pragma unroll
    for (int ct = 0; ct < 4; ++ct)
      s[ct] = (v8f){0.f,0.f,0.f,0.f,0.f,0.f,0.f,0.f};

    #pragma unroll
    for (int t = 0; t < 2; ++t) {
      #pragma unroll
      for (int ct = 0; ct < 4; ++ct) {
        const size_t kb = (size_t)(b * SEQ + n0 + ct * 16 + col) * D_MODEL
                          + h * HDIM + t * 32 + 8 * half;
        v16bf kH = ld_frag(Khi + kb, Khi + kb + 16);
        v16bf kL = ld_frag(Klo + kb, Klo + kb + 16);
        s[ct] = wmma_bf16(qH[t], kH, s[ct]);
        s[ct] = wmma_bf16(qH[t], kL, s[ct]);
        s[ct] = wmma_bf16(qL[t], kH, s[ct]);
      }
    }
    bool mk[4];
    #pragma unroll
    for (int ct = 0; ct < 4; ++ct)
      mk[ct] = amask[b * SEQ + n0 + ct * 16 + col] != 0;

    // ---- online softmax (row reductions within 16-lane halves) --------
    #pragma unroll
    for (int r = 0; r < 8; ++r) {
      float f[4];
      #pragma unroll
      for (int ct = 0; ct < 4; ++ct)
        f[ct] = mk[ct] ? s[ct][r] * scale : -1e30f;
      float mx = fmaxf(fmaxf(f[0], f[1]), fmaxf(f[2], f[3]));
      #pragma unroll
      for (int d = 8; d >= 1; d >>= 1) mx = fmaxf(mx, __shfl_xor(mx, d, 32));
      const float mn = fmaxf(mrun[r], mx);
      const float alpha = __expf(mrun[r] - mn);
      mrun[r] = mn;
      float p[4], rs = 0.f;
      #pragma unroll
      for (int ct = 0; ct < 4; ++ct) { p[ct] = __expf(f[ct] - mn); rs += p[ct]; }
      #pragma unroll
      for (int d = 8; d >= 1; d >>= 1) rs += __shfl_xor(rs, d, 32);
      lrun[r] = lrun[r] * alpha + rs;
      #pragma unroll
      for (int dt = 0; dt < 4; ++dt) oacc[dt][r] = oacc[dt][r] * alpha;

      const int prow = r + 8 * half;                 // C-layout row
      #pragma unroll
      for (int ct = 0; ct < 4; ++ct) {
        unsigned short ph = bf16_rn(p[ct]);
        ldsP[wave][0][prow * 64 + ct * 16 + col] = ph;
        ldsP[wave][1][prow * 64 + ct * 16 + col] = bf16_rn(p[ct] - bf16_f32(ph));
      }
    }

    // ---- O += P * V over two 32-deep k-steps ---------------------------
    // (same-wave DS ops are in-order; no barrier needed)
    #pragma unroll
    for (int t = 0; t < 2; ++t) {
      const unsigned short* pH0 = &ldsP[wave][0][col * 64 + t * 32 + 8 * half];
      const unsigned short* pL0 = &ldsP[wave][1][col * 64 + t * 32 + 8 * half];
      v16bf pH = ld_frag(pH0, pH0 + 16);
      v16bf pL = ld_frag(pL0, pL0 + 16);
      #pragma unroll
      for (int dt = 0; dt < 4; ++dt) {
        const size_t vo = ((size_t)((b * NHEAD + h) * HDIM + dt * 16 + col)) * SEQ
                          + n0 + t * 32 + 8 * half;
        v16bf vH = ld_frag(Vthi + vo, Vthi + vo + 16);
        v16bf vL = ld_frag(Vtlo + vo, Vtlo + vo + 16);
        oacc[dt] = wmma_bf16(pH, vH, oacc[dt]);
        oacc[dt] = wmma_bf16(pH, vL, oacc[dt]);
        oacc[dt] = wmma_bf16(pL, vH, oacc[dt]);
      }
    }
  }

  // ---- normalize and emit split O for the final projection ------------
  #pragma unroll
  for (int r = 0; r < 8; ++r) {
    const float inv = 1.0f / lrun[r];
    const size_t orow = (size_t)(b * SEQ + qt * 16 + r + 8 * half) * D_MODEL + h * HDIM;
    #pragma unroll
    for (int dt = 0; dt < 4; ++dt) {
      float v = oacc[dt][r] * inv;
      unsigned short hh = bf16_rn(v);
      Ohi[orow + dt * 16 + col] = hh;
      Olo[orow + dt * 16 + col] = bf16_rn(v - bf16_f32(hh));
    }
  }
}

// ---------------------------------------------------------------------------
extern "C" void kernel_launch(void* const* d_in, const int* in_sizes, int n_in,
                              void* d_out, int out_size, void* d_ws, size_t ws_size,
                              hipStream_t stream)
{
  (void)in_sizes; (void)n_in; (void)out_size; (void)ws_size;
  const float* x   = (const float*)d_in[0];
  const int*   amk = (const int*)  d_in[1];
  const float* Wq  = (const float*)d_in[2];
  const float* bq  = (const float*)d_in[3];
  const float* Wk  = (const float*)d_in[4];
  const float* bk  = (const float*)d_in[5];
  const float* Wv  = (const float*)d_in[6];
  const float* bv  = (const float*)d_in[7];
  const float* Wo  = (const float*)d_in[8];
  const float* bo  = (const float*)d_in[9];
  float* out = (float*)d_out;

  // Workspace carve-up (~80 MB): X planes are reused as O planes after QKV.
  char* ws = (char*)d_ws;
  size_t off = 0;
  auto alloc = [&](size_t bytes) -> unsigned short* {
    unsigned short* p = (unsigned short*)(ws + off);
    off += (bytes + 255) & ~(size_t)255;
    return p;
  };
  const size_t XB = (size_t)MTOT * D_MODEL * sizeof(unsigned short);    // 8 MB
  const size_t WB = (size_t)D_MODEL * D_MODEL * sizeof(unsigned short); // 2 MB
  unsigned short* Xhi  = alloc(XB);  // reused as Ohi
  unsigned short* Xlo  = alloc(XB);  // reused as Olo
  unsigned short* Wqhi = alloc(WB);  unsigned short* Wqlo = alloc(WB);
  unsigned short* Wkhi = alloc(WB);  unsigned short* Wklo = alloc(WB);
  unsigned short* Wvhi = alloc(WB);  unsigned short* Wvlo = alloc(WB);
  unsigned short* Wohi = alloc(WB);  unsigned short* Wolo = alloc(WB);
  unsigned short* Qhi  = alloc(XB);  unsigned short* Qlo  = alloc(XB);
  unsigned short* Khi  = alloc(XB);  unsigned short* Klo  = alloc(XB);
  unsigned short* Vthi = alloc(XB);  unsigned short* Vtlo = alloc(XB);

  // 1) split fp32 -> hi/lo bf16 planes
  {
    int n = MTOT * D_MODEL;
    split_kernel<<<(n + 255) / 256, 256, 0, stream>>>(x, Xhi, Xlo, n);
  }
  {
    int n = D_MODEL * D_MODEL;
    split_kernel<<<(n + 255) / 256, 256, 0, stream>>>(Wq, Wqhi, Wqlo, n);
    split_kernel<<<(n + 255) / 256, 256, 0, stream>>>(Wk, Wkhi, Wklo, n);
    split_kernel<<<(n + 255) / 256, 256, 0, stream>>>(Wv, Wvhi, Wvlo, n);
    split_kernel<<<(n + 255) / 256, 256, 0, stream>>>(Wo, Wohi, Wolo, n);
  }

  // 2) QKV projections (bias fused); V written pre-transposed per head
  dim3 g(MTOT / 64, D_MODEL / 128);
  gemm_bf16x3<0><<<g, 256, 0, stream>>>(Xhi, Xlo, Wqhi, Wqlo, bq, Qhi,  Qlo,  nullptr, D_MODEL, D_MODEL);
  gemm_bf16x3<0><<<g, 256, 0, stream>>>(Xhi, Xlo, Wkhi, Wklo, bk, Khi,  Klo,  nullptr, D_MODEL, D_MODEL);
  gemm_bf16x3<1><<<g, 256, 0, stream>>>(Xhi, Xlo, Wvhi, Wvlo, bv, Vthi, Vtlo, nullptr, D_MODEL, D_MODEL);

  // 3) flash attention: 4096 q-tiles total, 8 waves (q-tiles) per block
  flash_attn<<<(BATCH * NHEAD * (SEQ / 16)) / 8, 256, 0, stream>>>(
      Qhi, Qlo, Khi, Klo, Vthi, Vtlo, amk, /*O=*/Xhi, /*O=*/Xlo);

  // 4) output projection -> fp32 result
  gemm_bf16x3<2><<<g, 256, 0, stream>>>(Xhi, Xlo, Wohi, Wolo, bo, nullptr, nullptr, out, D_MODEL, D_MODEL);
}
